// ScaleDotProductAttention_15015205667534
// MI455X (gfx1250) — matrix operand order repaired and checked by reading, not compiled, not run on hardware
//
#include <hip/hip_runtime.h>
#include <hip/hip_bf16.h>

typedef __attribute__((ext_vector_type(16))) __bf16 v16bf;
typedef __attribute__((ext_vector_type(8)))  float  v8f;

union BF16Frag {
    v16bf v;
    unsigned int w[8];
    uint4 q[2];
};

constexpr int Bc = 2, Hc = 16, Sc = 2048, Dc = 128;
constexpr int BH = Bc * Hc;
constexpr float QK_SCALE = 0.08838834764831844f;  // 1/sqrt(128)

__device__ __forceinline__ unsigned int pack_bf16(float lo, float hi) {
    __bf16 a = (__bf16)lo, b = (__bf16)hi;
    unsigned short ua = __builtin_bit_cast(unsigned short, a);
    unsigned short ub = __builtin_bit_cast(unsigned short, b);
    return (unsigned int)ua | ((unsigned int)ub << 16);
}

// ---------------------------------------------------------------------------
// Elementwise f32 -> bf16 (row-major preserved). 8 elements per thread.
// ---------------------------------------------------------------------------
__global__ __launch_bounds__(256)
void cvt_bf16_kernel(const float* __restrict__ src, unsigned short* __restrict__ dst)
{
    size_t i = ((size_t)blockIdx.x * 256 + threadIdx.x) * 8;
    float4 a = *(const float4*)(src + i);
    float4 b = *(const float4*)(src + i + 4);
    uint4 o;
    o.x = pack_bf16(a.x, a.y);
    o.y = pack_bf16(a.z, a.w);
    o.z = pack_bf16(b.x, b.y);
    o.w = pack_bf16(b.z, b.w);
    *(uint4*)(dst + i) = o;
}

// ---------------------------------------------------------------------------
// V [bh][s][d] f32 -> vT [bh][d][s] bf16 via 32x32 LDS tile transpose.
// ---------------------------------------------------------------------------
__global__ __launch_bounds__(256)
void vtranspose_bf16_kernel(const float* __restrict__ v, unsigned short* __restrict__ vt)
{
    __shared__ unsigned short tile[32][33];
    const int bh = blockIdx.z;
    const int s0 = blockIdx.x * 32;
    const int d0 = blockIdx.y * 32;
    const int tx = threadIdx.x & 31;
    const int ty = threadIdx.x >> 5;    // 0..7

    const float* vb = v + (size_t)bh * Sc * Dc;
    #pragma unroll
    for (int i = 0; i < 4; ++i) {
        const int s = ty + 8 * i;
        __bf16 h = (__bf16)vb[(size_t)(s0 + s) * Dc + d0 + tx];
        tile[s][tx] = __builtin_bit_cast(unsigned short, h);
    }
    __syncthreads();
    unsigned short* vo = vt + (size_t)bh * Dc * Sc;
    #pragma unroll
    for (int i = 0; i < 4; ++i) {
        const int d = ty + 8 * i;
        vo[(size_t)(d0 + d) * Sc + s0 + tx] = tile[tx][d];
    }
}

// ---------------------------------------------------------------------------
// Pass 1: raw scores = mask(QK^T * scale) -> attn (raw), per-row online
// softmax stats (max, sum-of-exp) -> d_ws. One wave per 16-row Q tile.
// All fragments are direct b128 loads of pre-converted bf16.
// ---------------------------------------------------------------------------
__global__ __launch_bounds__(256)
void fa_scores_kernel(const unsigned short* __restrict__ qB,
                      const unsigned short* __restrict__ kB,
                      const int* __restrict__ mask,
                      float* __restrict__ attn, float2* __restrict__ stats)
{
    const int lane = threadIdx.x & 31;
    const int wave = threadIdx.x >> 5;
    const int mt   = blockIdx.x * 8 + wave;
    const int bh   = mt >> 7;                 // 128 m-tiles per (b,h)
    const int m0   = (mt & 127) << 4;

    const unsigned short* qb = qB + (size_t)bh * Sc * Dc;
    const unsigned short* kb = kB + (size_t)bh * Sc * Dc;
    float* attnb = attn + (size_t)bh * Sc * Sc;

    const int half  = lane >> 4;
    const int lid   = lane & 15;
    const int koffA = half ? 8 : 0;           // A-frag interleave (ISA 7.12.2)
    const int koffB = half ? 8 : 0;           // B-frag interleave (matches A)

    // Q tile A-fragments: 4 chunks of K=32, two 16B loads each.
    BF16Frag qa[4];
    {
        const unsigned short* qrow = qb + (size_t)(m0 + lid) * Dc;
        #pragma unroll
        for (int c = 0; c < 4; ++c) {
            const unsigned short* p0 = qrow + c * 32 + koffA;
            qa[c].q[0] = *(const uint4*)(p0);        // K = koffA..koffA+7
            qa[c].q[1] = *(const uint4*)(p0 + 16);   // K = 16+koffA..16+koffA+7
        }
    }

    float rm[8], rs[8];
    #pragma unroll
    for (int j = 0; j < 8; ++j) { rm[j] = -3.0e38f; rs[j] = 0.0f; }

    for (int nt = 0; nt < Sc / 16; ++nt) {
        const int n0 = nt << 4;
        const unsigned short* krow = kb + (size_t)(n0 + lid) * Dc;

        v8f c = {};
        #pragma unroll
        for (int kc = 0; kc < 4; ++kc) {
            BF16Frag bf;
            const unsigned short* p0 = krow + kc * 32 + koffB;
            bf.q[0] = *(const uint4*)(p0);           // K = koffB..koffB+7
            bf.q[1] = *(const uint4*)(p0 + 16);      // K = 16+koffB..16+koffB+7
            c = __builtin_amdgcn_wmma_f32_16x16x32_bf16(
                    false, qa[kc].v, false, bf.v, (short)0, c, false, false);
        }

        #pragma unroll
        for (int j = 0; j < 8; ++j) {
            const int mg = m0 + (half << 3) + j;
            const int ng = n0 + lid;
            float s = c[j] * QK_SCALE;
            if (mask[(size_t)mg * Sc + ng] == 0) s = -1.0e10f;
            attnb[(size_t)mg * Sc + ng] = s;
            float nm = fmaxf(rm[j], s);
            rs[j] = rs[j] * __expf(rm[j] - nm) + __expf(s - nm);
            rm[j] = nm;
        }
    }

    #pragma unroll
    for (int j = 0; j < 8; ++j) {
        #pragma unroll
        for (int xm = 1; xm < 16; xm <<= 1) {
            float om = __shfl_xor(rm[j], xm, 32);
            float os = __shfl_xor(rs[j], xm, 32);
            float nm = fmaxf(rm[j], om);
            rs[j] = rs[j] * __expf(rm[j] - nm) + os * __expf(om - nm);
            rm[j] = nm;
        }
    }
    if (lid == 0) {
        #pragma unroll
        for (int j = 0; j < 8; ++j) {
            const int row = m0 + (half << 3) + j;
            stats[(size_t)bh * Sc + row] = make_float2(rm[j], rs[j]);
        }
    }
}

// ---------------------------------------------------------------------------
// Pass 2: normalize scores (exp(s-max)/sum) -> attn in place; out = P @ V.
// V^T chunks staged into LDS with GLOBAL_LOAD_ASYNC_TO_LDS_B128 (ASYNCcnt),
// double-buffered so chunk kt+1 streams in while chunk kt feeds the WMMAs.
// B-fragments are two conflict-free ds_load_b128 (stride-20-word padding).
// ---------------------------------------------------------------------------
__global__ __launch_bounds__(256)
void fa_pv_kernel(const unsigned short* __restrict__ vTb, float* __restrict__ attn,
                  const float2* __restrict__ stats, float* __restrict__ out)
{
    __shared__ unsigned int vT[2][Dc * 20];   // double buffer, stride 20 words/col

    const int lane = threadIdx.x & 31;
    const int wave = threadIdx.x >> 5;
    const int mt   = blockIdx.x * 8 + wave;
    const int bh   = mt >> 7;
    const int m0   = (mt & 127) << 4;

    const unsigned short* vtb = vTb + (size_t)bh * Dc * Sc;
    float* attnb = attn + (size_t)bh * Sc * Sc;
    float* outb  = out + (size_t)bh * Sc * Dc;

    const int half  = lane >> 4;
    const int lid   = lane & 15;
    const int koffA = half ? 8 : 0;
    const int bOff  = half ? 4 : 0;        // B-frag kd-pair offset (kd base 8, interleaved)

    const float2 st = stats[(size_t)bh * Sc + m0 + lid];
    const float rowMax = st.x;
    const float rowInv = 1.0f / st.y;

    float* srow = attnb + (size_t)(m0 + lid) * Sc;

    v8f acc[8] = {};

    // staging assignment: thread -> (column, 16-element half of the 32 kd's)
    const int sCol = threadIdx.x >> 1;     // 0..127
    const int sH   = threadIdx.x & 1;      // 0 or 1
    const unsigned int ldsOff = (unsigned int)((sCol * 20 + sH * 8) * 4);

    // Async global->LDS copy of one 32-wide V^T chunk slice (2 x 16B).
    // INST_OFFSET is added to both LDS and global addresses (ISA 10.x), so the
    // second chunk is the same operands with offset:16.
    auto stage = [&](int buf, int kb0) {
        const unsigned short* src = vtb + (size_t)sCol * Sc + kb0 + sH * 16;
        unsigned long long ga = (unsigned long long)(uintptr_t)src;
        unsigned int l0 = ldsOff + (unsigned int)(buf * (Dc * 20 * 4));
        asm volatile("global_load_async_to_lds_b128 %0, %1, off"
                     :: "v"(l0), "v"(ga) : "memory");
        asm volatile("global_load_async_to_lds_b128 %0, %1, off offset:16"
                     :: "v"(l0), "v"(ga) : "memory");
    };

    constexpr int NT = Sc / 32;   // 64
    stage(0, 0);

    for (int kt = 0; kt < NT; ++kt) {
        const int cur = kt & 1;
        const int kb0 = kt * 32;

        if (kt + 1 < NT) {
            stage(1 - cur, kb0 + 32);   // prefetch next chunk into other buffer
            asm volatile("s_wait_asynccnt 0x2" ::: "memory");  // cur's 2 done
        } else {
            asm volatile("s_wait_asynccnt 0x0" ::: "memory");
        }
        __syncthreads();   // all waves' cur-chunk async loads complete & visible

        // A fragment: raw scores -> p = exp(s-max)/sum; write attn back.
        BF16Frag a;
        {
            float* s0 = srow + kb0 + koffA;
            float4 f0 = *(const float4*)(s0);
            float4 f1 = *(const float4*)(s0 + 4);
            float4 f2 = *(const float4*)(s0 + 16);
            float4 f3 = *(const float4*)(s0 + 20);
            float p[16] = {f0.x, f0.y, f0.z, f0.w, f1.x, f1.y, f1.z, f1.w,
                           f2.x, f2.y, f2.z, f2.w, f3.x, f3.y, f3.z, f3.w};
            #pragma unroll
            for (int i = 0; i < 16; ++i)
                p[i] = __expf(p[i] - rowMax) * rowInv;
            *(float4*)(s0)      = make_float4(p[0],  p[1],  p[2],  p[3]);
            *(float4*)(s0 + 4)  = make_float4(p[4],  p[5],  p[6],  p[7]);
            *(float4*)(s0 + 16) = make_float4(p[8],  p[9],  p[10], p[11]);
            *(float4*)(s0 + 20) = make_float4(p[12], p[13], p[14], p[15]);
            #pragma unroll
            for (int i = 0; i < 16; i += 2)
                a.w[i >> 1] = pack_bf16(p[i], p[i + 1]);
        }

        const unsigned int* vbuf = vT[cur];
        #pragma unroll
        for (int nt = 0; nt < 8; ++nt) {
            const int col = nt * 16 + lid;
            const uint4* bp = (const uint4*)&vbuf[col * 20 + bOff];
            BF16Frag b;
            b.q[0] = bp[0];
            b.q[1] = bp[2];
            acc[nt] = __builtin_amdgcn_wmma_f32_16x16x32_bf16(
                          false, a.v, false, b.v, (short)0, acc[nt], false, false);
        }

        __syncthreads();   // done with cur buffer before it is re-staged (kt+2)
    }

    #pragma unroll
    for (int nt = 0; nt < 8; ++nt) {
        #pragma unroll
        for (int j = 0; j < 8; ++j) {
            const int mg = m0 + (half << 3) + j;
            outb[(size_t)mg * Dc + nt * 16 + lid] = acc[nt][j];
        }
    }
}

extern "C" void kernel_launch(void* const* d_in, const int* in_sizes, int n_in,
                              void* d_out, int out_size, void* d_ws, size_t ws_size,
                              hipStream_t stream)
{
    (void)in_sizes; (void)n_in; (void)out_size; (void)ws_size;
    const float* q  = (const float*)d_in[0];
    const float* k  = (const float*)d_in[1];
    const float* v  = (const float*)d_in[2];
    const int* mask = (const int*)d_in[3];

    float* out  = (float*)d_out;                       // [B,H,S,D]
    float* attn = out + (size_t)BH * Sc * Dc;          // [B,H,S,S]

    // Workspace layout: stats (512 KB) | qB | kB | vT  (bf16, 16.8 MB each)
    char* ws = (char*)d_ws;
    float2* stats        = (float2*)ws;
    unsigned short* qB   = (unsigned short*)(ws + (size_t)512 * 1024);
    unsigned short* kB   = qB + (size_t)BH * Sc * Dc;
    unsigned short* vTb  = kB + (size_t)BH * Sc * Dc;

    const int nElem   = BH * Sc * Dc;                  // 8,388,608
    const int cvtBlks = nElem / (256 * 8);             // 4096

    cvt_bf16_kernel<<<cvtBlks, 256, 0, stream>>>(q, qB);
    cvt_bf16_kernel<<<cvtBlks, 256, 0, stream>>>(k, kB);
    dim3 tgrid(Sc / 32, Dc / 32, BH);
    vtranspose_bf16_kernel<<<tgrid, 256, 0, stream>>>(v, vTb);

    const int blocks = (BH * Sc / 16) / 8;             // 512 blocks x 256 threads
    fa_scores_kernel<<<blocks, 256, 0, stream>>>(qB, kB, mask, attn, stats);
    fa_pv_kernel<<<blocks, 256, 0, stream>>>(vTb, attn, stats, out);
}
